// SwinSR_31791347925725
// MI455X (gfx1250) — compile-verified, matrix operand order checked by reading, not run on hardware
//
#include <hip/hip_runtime.h>
#include <hip/hip_bf16.h>
#include <math.h>

typedef __attribute__((ext_vector_type(16))) _Float16 v16h;
typedef __attribute__((ext_vector_type(8)))  _Float16 v8h;
typedef __attribute__((ext_vector_type(4)))  _Float16 v4h;
typedef __attribute__((ext_vector_type(8)))  float    v8f;

constexpr int TOKENS = 25088;   // 8 * 56 * 56

constexpr int EPI_BIAS_F32   = 0;
constexpr int EPI_BIAS_RES   = 1;
constexpr int EPI_GELU_F16   = 2;
constexpr int EPI_SHUF_PRELU = 3;

// Low 32 bits of the flat address of an LDS object == LDS byte offset
__device__ inline unsigned lds_off_u32(const void* p) {
  return (unsigned)(unsigned long long)p;
}

// ---------------------------------------------------------------------------
// WMMA GEMM: C[M,N] = A[M,K](f16) * Bw[N,K]^T(f16) + epilogue.
// Tile: 256M x 32N per workgroup (8 waves), K-step 32.
// Each wave: 32M x 32N = 2 A-frags x 2 B-frags -> 4 v_wmma_f32_16x16x32_f16.
// AMODE 0: A row-major [M,K]; tiles staged with GLOBAL_LOAD_ASYNC_TO_LDS
//          (ASYNCcnt) into double-buffered LDS: next tile's DMA overlaps
//          current tile's WMMAs, gated by s_wait_asynccnt 5.
// AMODE 1: A = feat16 [8,56,56,96], im2col 3x3 gather (k = c*9 + dy*3 + dx,
//          matching up_w [O][I][3][3] flattening); scalar path.
// ---------------------------------------------------------------------------
template<int AMODE, int EPI>
__global__ __launch_bounds__(256) void gemm_wmma_kernel(
    const _Float16* __restrict__ A, const _Float16* __restrict__ Bw,
    const float* __restrict__ bias, float* __restrict__ out32,
    _Float16* __restrict__ out16, const float* __restrict__ alpha,
    int M, int N, int K)
{
  constexpr int BM = 256, BN = 32, BK = 32;
  __shared__ alignas(16) _Float16 As[2][BM * BK];
  __shared__ alignas(16) _Float16 Bs[2][BN * BK];

  const int tid   = threadIdx.x;
  const int lane  = tid & 31;
  const int wv    = tid >> 5;
  const int row16 = lane & 15;
  const int g     = (lane >> 4) & 1;
  const int m0 = blockIdx.x * BM;
  const int n0 = blockIdx.y * BN;

  v8f acc[2][2];
  #pragma unroll
  for (int a = 0; a < 2; ++a)
    #pragma unroll
    for (int b = 0; b < 2; ++b)
      #pragma unroll
      for (int i = 0; i < 8; ++i) acc[a][b][i] = 0.f;

  // B-tile cooperative mapping: thread -> 4 contiguous halfs (8 B).
  const int be_n = (tid << 2) >> 5;   // 0..31
  const int be_k = (tid << 2) & 31;

  // Fragment assembly + 4 WMMAs for one LDS buffer.
  auto compute_tile = [&](int buf) {
    v16h afr[2], bfr[2];
    #pragma unroll
    for (int mi = 0; mi < 2; ++mi) {
      const _Float16* p = &As[buf][(wv * 32 + mi * 16 + row16) * BK];
      v8h lo = *(const v8h*)(p + g * 8);
      v8h hi = *(const v8h*)(p + 16 + g * 8);
      #pragma unroll
      for (int i = 0; i < 8; ++i) { afr[mi][i] = lo[i]; afr[mi][i + 8] = hi[i]; }
    }
    #pragma unroll
    for (int ni = 0; ni < 2; ++ni) {
      const _Float16* p = &Bs[buf][(ni * 16 + row16) * BK];
      v8h lo = *(const v8h*)(p + g * 8);
      v8h hi = *(const v8h*)(p + 16 + g * 8);
      #pragma unroll
      for (int i = 0; i < 8; ++i) { bfr[ni][i] = lo[i]; bfr[ni][i + 8] = hi[i]; }
    }
    #pragma unroll
    for (int mi = 0; mi < 2; ++mi)
      #pragma unroll
      for (int ni = 0; ni < 2; ++ni)
        acc[mi][ni] = __builtin_amdgcn_wmma_f32_16x16x32_f16(
            false, afr[mi], false, bfr[ni], (short)0, acc[mi][ni], false, false);
  };

  if constexpr (AMODE == 0) {
    // Issue the 5 async DMA ops (4x b128 A-row + 1x b64 B-chunk) for one tile.
    auto issue_tile = [&](int k0, int buf) {
      unsigned ldsA = lds_off_u32(&As[buf][tid * BK]);
      unsigned long long gA =
          (unsigned long long)(const void*)(A + (size_t)(m0 + tid) * K + k0);
      asm volatile("global_load_async_to_lds_b128 %0, %1, off"
                   :: "v"(ldsA), "v"(gA) : "memory");
      asm volatile("global_load_async_to_lds_b128 %0, %1, off offset:16"
                   :: "v"(ldsA), "v"(gA) : "memory");
      asm volatile("global_load_async_to_lds_b128 %0, %1, off offset:32"
                   :: "v"(ldsA), "v"(gA) : "memory");
      asm volatile("global_load_async_to_lds_b128 %0, %1, off offset:48"
                   :: "v"(ldsA), "v"(gA) : "memory");
      unsigned ldsB = lds_off_u32(&Bs[buf][be_n * BK + be_k]);
      unsigned long long gB =
          (unsigned long long)(const void*)(Bw + (size_t)(n0 + be_n) * K + k0 + be_k);
      asm volatile("global_load_async_to_lds_b64 %0, %1, off"
                   :: "v"(ldsB), "v"(gB) : "memory");
    };

    const int KT = K / BK;
    issue_tile(0, 0);
    for (int kt = 0; kt < KT; ++kt) {
      int buf = kt & 1;
      if (kt + 1 < KT) {
        issue_tile((kt + 1) * BK, buf ^ 1);     // prefetch next tile (5 ops in flight)
        asm volatile("s_wait_asynccnt 0x5" ::: "memory");  // current tile landed
      } else {
        asm volatile("s_wait_asynccnt 0x0" ::: "memory");
      }
      __syncthreads();
      compute_tile(buf);
      __syncthreads();                          // LDS reuse fence for buf^1 rewrite
    }
  } else {
    int m = m0 + tid;
    int gb_b = m / 3136, rem = m % 3136, gb_h = rem / 56, gb_w = rem % 56;
    for (int k0 = 0; k0 < K; k0 += BK) {
      #pragma unroll 4
      for (int kl = 0; kl < BK; ++kl) {
        int k = k0 + kl;
        int c = k / 9, q = k % 9;
        int yy = gb_h + q / 3 - 1;
        int xx = gb_w + q % 3 - 1;
        _Float16 v = (_Float16)0.f;
        if ((unsigned)yy < 56u && (unsigned)xx < 56u)
          v = A[((size_t)((gb_b * 56 + yy) * 56 + xx)) * 96 + c];
        As[0][tid * BK + kl] = v;
      }
      *(v4h*)(&Bs[0][be_n * BK + be_k]) =
          *(const v4h*)(Bw + (size_t)(n0 + be_n) * K + k0 + be_k);
      __syncthreads();
      compute_tile(0);
      __syncthreads();
    }
  }

  // ---- epilogue: C layout lane=N, vgpr r -> M = r + 8*(lane>>4) ----
  float al = 0.f;
  if constexpr (EPI == EPI_SHUF_PRELU) al = alpha[0];
  #pragma unroll
  for (int mi = 0; mi < 2; ++mi)
    #pragma unroll
    for (int ni = 0; ni < 2; ++ni)
      #pragma unroll
      for (int r = 0; r < 8; ++r) {
        int m = m0 + wv * 32 + mi * 16 + r + (g << 3);
        int n = n0 + ni * 16 + row16;
        float v = acc[mi][ni][r] + bias[n];
        if constexpr (EPI == EPI_BIAS_F32) {
          out32[(size_t)m * N + n] = v;
        } else if constexpr (EPI == EPI_BIAS_RES) {
          out32[(size_t)m * N + n] = v + out32[(size_t)m * N + n];
        } else if constexpr (EPI == EPI_GELU_F16) {
          out16[(size_t)m * N + n] = (_Float16)(0.5f * v * (1.f + erff(v * 0.70710678f)));
        } else {
          // fused pixel-shuffle (r=4) + PReLU; out16 = up [8][96][224][224]
          int b = m / 3136, rem = m % 3136, hh = rem / 56, ww = rem % 56;
          int co = n >> 4, ii = (n >> 2) & 3, jj = n & 3;
          float pv = v >= 0.f ? v : al * v;
          out16[(((size_t)b * 96 + co) * 224 + (hh * 4 + ii)) * 224 + (ww * 4 + jj)] =
              (_Float16)pv;
        }
      }
}

// ---------------------------------------------------------------------------
// LayerNorm over C=96, one wave per token (3 channels per lane, wave32 xor
// reduction). Optionally writes f32 (residual stream) and/or f16 (GEMM A).
// ---------------------------------------------------------------------------
__global__ __launch_bounds__(256) void ln96_kernel(
    const float* __restrict__ in, const float* __restrict__ gma,
    const float* __restrict__ bta, float* __restrict__ out32,
    _Float16* __restrict__ out16, int M)
{
  int wv = threadIdx.x >> 5, lane = threadIdx.x & 31;
  int t = blockIdx.x * 8 + wv;
  if (t >= M) return;
  const float* row = in + (size_t)t * 96;
  float v0 = row[lane], v1 = row[lane + 32], v2 = row[lane + 64];
  float s = v0 + v1 + v2;
  #pragma unroll
  for (int o = 16; o > 0; o >>= 1) s += __shfl_xor(s, o, 32);
  float mean = s * (1.f / 96.f);
  float d0 = v0 - mean, d1 = v1 - mean, d2 = v2 - mean;
  float q = d0 * d0 + d1 * d1 + d2 * d2;
  #pragma unroll
  for (int o = 16; o > 0; o >>= 1) q += __shfl_xor(q, o, 32);
  float rstd = rsqrtf(q * (1.f / 96.f) + 1e-5f);
  float r0 = d0 * rstd * gma[lane]      + bta[lane];
  float r1 = d1 * rstd * gma[lane + 32] + bta[lane + 32];
  float r2 = d2 * rstd * gma[lane + 64] + bta[lane + 64];
  size_t o0 = (size_t)t * 96 + lane;
  if (out32) { out32[o0] = r0; out32[o0 + 32] = r1; out32[o0 + 64] = r2; }
  if (out16) { out16[o0] = (_Float16)r0; out16[o0 + 32] = (_Float16)r1; out16[o0 + 64] = (_Float16)r2; }
}

// ---------------------------------------------------------------------------
// Window attention (N=49, hd = 96/heads in {32,16,8,4}).
// One workgroup per (window, head); thread n owns query row n.
// Shift handled by gathering qkv rows at rolled token index and scattering
// the output back to the same index (roll + inverse roll compose).
// Relative-position bias and cyclic-shift mask computed arithmetically.
// ---------------------------------------------------------------------------
__global__ __launch_bounds__(64) void win_attn_kernel(
    const float* __restrict__ qkv, const float* __restrict__ rpb,
    _Float16* __restrict__ outh, int heads, int shift)
{
  __shared__ float qs[49 * 32];
  __shared__ float ks[49 * 32];
  __shared__ float vs[49 * 32];
  __shared__ float ps[49 * 49];
  __shared__ int   rs[49];
  const int hd = 96 / heads;
  const int w = blockIdx.x;            // 0..511
  const int head = blockIdx.y;
  const int b = w >> 6, wi = w & 63, wy = wi >> 3, wx = wi & 7;
  const int n = threadIdx.x;
  int t_n = 0, myreg = 0;
  if (n < 49) {
    int r = n / 7, c = n % 7;
    int ys = wy * 7 + r, xs = wx * 7 + c;              // shifted-space coords
    int y = (ys + shift) % 56, x = (xs + shift) % 56;  // original token
    t_n = (b * 56 + y) * 56 + x;
    const float* base = qkv + (size_t)t_n * 288 + head * hd;
    float sc = rsqrtf((float)hd);
    for (int d = 0; d < hd; ++d) {
      qs[n * 32 + d] = base[d] * sc;
      ks[n * 32 + d] = base[96 + d];
      vs[n * 32 + d] = base[192 + d];
    }
    int ry = ys < 49 ? 0 : (ys < 53 ? 1 : 2);          // H-7=49, H-3=53
    int rx = xs < 49 ? 0 : (xs < 53 ? 1 : 2);
    myreg = ry * 3 + rx;
    rs[n] = myreg;
  }
  __syncthreads();
  if (n >= 49) return;
  int r1 = n / 7, c1 = n % 7;
  float mx = -1e30f;
  for (int m = 0; m < 49; ++m) {
    float s = 0.f;
    for (int d = 0; d < hd; ++d) s += qs[n * 32 + d] * ks[m * 32 + d];
    int r2 = m / 7, c2 = m % 7;
    s += rpb[((r1 - r2 + 6) * 13 + (c1 - c2 + 6)) * heads + head];
    if (shift && rs[m] != myreg) s -= 100.f;
    ps[n * 49 + m] = s;
    mx = fmaxf(mx, s);
  }
  float sum = 0.f;
  for (int m = 0; m < 49; ++m) {
    float e = expf(ps[n * 49 + m] - mx);
    ps[n * 49 + m] = e;
    sum += e;
  }
  float inv = 1.f / sum;
  for (int d = 0; d < hd; ++d) {
    float o = 0.f;
    for (int m = 0; m < 49; ++m) o += ps[n * 49 + m] * vs[m * 32 + d];
    outh[(size_t)t_n * 96 + head * hd + d] = (_Float16)(o * inv);
  }
}

// ---------------------------------------------------------------------------
// Patch embed: 4x4 s4 conv, thread per (token, channel). K=48 only.
// ---------------------------------------------------------------------------
__global__ __launch_bounds__(256) void patch_embed_kernel(
    const float* __restrict__ x, const float* __restrict__ w,
    const float* __restrict__ bias, float* __restrict__ out)
{
  int idx = blockIdx.x * 256 + threadIdx.x;
  if (idx >= TOKENS * 96) return;
  int c = idx % 96; int t = idx / 96;
  int b = t / 3136, rem = t % 3136, y = rem / 56, xx = rem % 56;
  float acc = bias[c];
  for (int ci = 0; ci < 3; ++ci)
    #pragma unroll
    for (int ky = 0; ky < 4; ++ky)
      #pragma unroll
      for (int kx = 0; kx < 4; ++kx)
        acc += x[((size_t)(b * 3 + ci) * 224 + (y * 4 + ky)) * 224 + (xx * 4 + kx)]
             * w[((c * 3 + ci) * 4 + ky) * 4 + kx];
  out[(size_t)t * 96 + c] = acc;
}

// ---------------------------------------------------------------------------
// Final 3x3 conv to RGB. up already has PReLU applied.
// ---------------------------------------------------------------------------
__global__ __launch_bounds__(256) void rc_conv_kernel(
    const _Float16* __restrict__ up, const float* __restrict__ w,
    const float* __restrict__ bias, float* __restrict__ out)
{
  int idx = blockIdx.x * 256 + threadIdx.x;
  if (idx >= 8 * 3 * 224 * 224) return;
  int X = idx % 224; int t = idx / 224;
  int Y = t % 224; t /= 224;
  int cc = t % 3; int b = t / 3;
  float acc = bias[cc];
  for (int cu = 0; cu < 96; ++cu) {
    const float* wr = w + (cc * 96 + cu) * 9;
    const _Float16* ub = up + ((size_t)b * 96 + cu) * 224 * 224;
    #pragma unroll
    for (int dy = 0; dy < 3; ++dy) {
      int yy = Y + dy - 1;
      if ((unsigned)yy >= 224u) continue;
      #pragma unroll
      for (int dx = 0; dx < 3; ++dx) {
        int xx = X + dx - 1;
        if ((unsigned)xx >= 224u) continue;
        acc += wr[dy * 3 + dx] * (float)ub[yy * 224 + xx];
      }
    }
  }
  out[idx] = acc;
}

__global__ __launch_bounds__(256) void f32_to_f16_kernel(
    const float* __restrict__ s, _Float16* __restrict__ d, int n)
{
  int i = blockIdx.x * 256 + threadIdx.x;
  if (i < n) d[i] = (_Float16)s[i];
}

// ---------------------------------------------------------------------------
extern "C" void kernel_launch(void* const* d_in, const int* in_sizes, int n_in,
                              void* d_out, int out_size, void* d_ws, size_t ws_size,
                              hipStream_t stream) {
  (void)in_sizes; (void)n_in; (void)out_size; (void)ws_size;
  const float* x_in    = (const float*)d_in[0];
  const float* pe_w    = (const float*)d_in[1];
  const float* pe_b    = (const float*)d_in[2];
  const float* pe_ln_g = (const float*)d_in[3];
  const float* pe_ln_b = (const float*)d_in[4];
  const float* norm_g  = (const float*)d_in[5];
  const float* norm_b  = (const float*)d_in[6];
  const float* up_w    = (const float*)d_in[7];
  const float* up_b    = (const float*)d_in[8];
  const float* prelu_a = (const float*)d_in[9];
  const float* rc_w    = (const float*)d_in[10];
  const float* rc_b    = (const float*)d_in[11];

  char* wsp = (char*)d_ws;
  size_t off = 0;
  auto alloc = [&](size_t bytes) {
    size_t r = off; off += (bytes + 255) & ~(size_t)255; return r;
  };

  float*    xbuf   = (float*)   (wsp + alloc((size_t)TOKENS * 96 * 4));
  float*    petmp  = (float*)   (wsp + alloc((size_t)TOKENS * 96 * 4));
  float*    qkvb   = (float*)   (wsp + alloc((size_t)TOKENS * 288 * 4));
  _Float16* h16    = (_Float16*)(wsp + alloc((size_t)TOKENS * 96 * 2));
  _Float16* attn16 = (_Float16*)(wsp + alloc((size_t)TOKENS * 96 * 2));
  _Float16* mlp16  = (_Float16*)(wsp + alloc((size_t)TOKENS * 384 * 2));
  _Float16* feat16 = (_Float16*)(wsp + alloc((size_t)TOKENS * 96 * 2));
  _Float16* up16   = (_Float16*)(wsp + alloc((size_t)8 * 96 * 224 * 224 * 2));
  _Float16* wup16  = (_Float16*)(wsp + alloc((size_t)1536 * 864 * 2));
  _Float16 *wqkv16[12], *wproj16[12], *wfc116[12], *wfc216[12];
  for (int i = 0; i < 12; ++i) {
    wqkv16[i]  = (_Float16*)(wsp + alloc(288 * 96 * 2));
    wproj16[i] = (_Float16*)(wsp + alloc(96 * 96 * 2));
    wfc116[i]  = (_Float16*)(wsp + alloc(384 * 96 * 2));
    wfc216[i]  = (_Float16*)(wsp + alloc(96 * 384 * 2));
  }

  auto cvt = [&](const float* s, _Float16* d, int n) {
    f32_to_f16_kernel<<<(n + 255) / 256, 256, 0, stream>>>(s, d, n);
  };

  static const int HEADS_CFG[12] = {3, 3, 6, 6, 12, 12, 12, 12, 12, 12, 24, 24};
  static const int SHIFT_CFG[12] = {0, 3, 0, 3, 0, 3, 0, 3, 0, 3, 0, 3};

  for (int blk = 0; blk < 12; ++blk) {
    int base = 12 + blk * 13;
    cvt((const float*)d_in[base + 2],  wqkv16[blk],  288 * 96);
    cvt((const float*)d_in[base + 4],  wproj16[blk], 96 * 96);
    cvt((const float*)d_in[base + 9],  wfc116[blk],  384 * 96);
    cvt((const float*)d_in[base + 11], wfc216[blk],  96 * 384);
  }
  cvt(up_w, wup16, 1536 * 864);

  patch_embed_kernel<<<(TOKENS * 96 + 255) / 256, 256, 0, stream>>>(x_in, pe_w, pe_b, petmp);
  ln96_kernel<<<TOKENS / 8, 256, 0, stream>>>(petmp, pe_ln_g, pe_ln_b, xbuf, nullptr, TOKENS);

  for (int blk = 0; blk < 12; ++blk) {
    int base = 12 + blk * 13;
    const float* ln1_g = (const float*)d_in[base + 0];
    const float* ln1_b = (const float*)d_in[base + 1];
    const float* qkv_b = (const float*)d_in[base + 3];
    const float* proj_b= (const float*)d_in[base + 5];
    const float* rpb   = (const float*)d_in[base + 6];
    const float* ln2_g = (const float*)d_in[base + 7];
    const float* ln2_b = (const float*)d_in[base + 8];
    const float* fc1_b = (const float*)d_in[base + 10];
    const float* fc2_b = (const float*)d_in[base + 12];
    int heads = HEADS_CFG[blk], shift = SHIFT_CFG[blk];

    ln96_kernel<<<TOKENS / 8, 256, 0, stream>>>(xbuf, ln1_g, ln1_b, nullptr, h16, TOKENS);
    gemm_wmma_kernel<0, EPI_BIAS_F32><<<dim3(TOKENS / 256, 288 / 32), 256, 0, stream>>>(
        h16, wqkv16[blk], qkv_b, qkvb, nullptr, nullptr, TOKENS, 288, 96);
    win_attn_kernel<<<dim3(512, heads), 64, 0, stream>>>(qkvb, rpb, attn16, heads, shift);
    gemm_wmma_kernel<0, EPI_BIAS_RES><<<dim3(TOKENS / 256, 96 / 32), 256, 0, stream>>>(
        attn16, wproj16[blk], proj_b, xbuf, nullptr, nullptr, TOKENS, 96, 96);
    ln96_kernel<<<TOKENS / 8, 256, 0, stream>>>(xbuf, ln2_g, ln2_b, nullptr, h16, TOKENS);
    gemm_wmma_kernel<0, EPI_GELU_F16><<<dim3(TOKENS / 256, 384 / 32), 256, 0, stream>>>(
        h16, wfc116[blk], fc1_b, nullptr, mlp16, nullptr, TOKENS, 384, 96);
    gemm_wmma_kernel<0, EPI_BIAS_RES><<<dim3(TOKENS / 256, 96 / 32), 256, 0, stream>>>(
        mlp16, wfc216[blk], fc2_b, xbuf, nullptr, nullptr, TOKENS, 96, 384);
  }

  ln96_kernel<<<TOKENS / 8, 256, 0, stream>>>(xbuf, norm_g, norm_b, nullptr, feat16, TOKENS);
  // up-conv as implicit GEMM with fused pixel-shuffle + PReLU (66 GFLOP, the big one)
  gemm_wmma_kernel<1, EPI_SHUF_PRELU><<<dim3(TOKENS / 256, 1536 / 32), 256, 0, stream>>>(
      feat16, wup16, up_b, nullptr, up16, prelu_a, TOKENS, 1536, 864);
  rc_conv_kernel<<<(8 * 3 * 224 * 224 + 255) / 256, 256, 0, stream>>>(
      up16, rc_w, rc_b, (float*)d_out);
}
